// ImplicitNetOC_9964324127525
// MI455X (gfx1250) — matrix operand order repaired and checked by reading, not compiled, not run on hardware
//
#include <hip/hip_runtime.h>
#include <hip/hip_bf16.h>

// ---------------------------------------------------------------------------
// Fused implicit-OC network for MI455X (gfx1250, wave32, WMMA bf16)
// ---------------------------------------------------------------------------

typedef __attribute__((ext_vector_type(16))) __bf16 v16bf;
typedef __attribute__((ext_vector_type(2)))  __bf16 v2bf;
typedef __attribute__((ext_vector_type(8)))  float  v8f;

#define STATE_DIM   64
#define CONTROL_DIM 16
#define HIDDEN      256
#define MTILE       128          // batch rows per workgroup
#define THREADS     256          // 8 waves (wave32)
#define HPAD        264          // LDS row stride (bf16 elems) = 528B, bank-friendly

union V16U { uint4 u[2]; v16bf v; };

__device__ __forceinline__ __bf16 f2bf(float f) {
    return (__bf16)f;                           // native v_cvt on gfx1250
}
__device__ __forceinline__ unsigned short f2bf_bits(float f) {
    return __builtin_bit_cast(unsigned short, (__bf16)f);
}
__device__ __forceinline__ float relu1(float v) {
    // single v_med3_f32: median(v, 0, +inf) == max(v, 0)
    return __builtin_amdgcn_fmed3f(v, 0.0f, __builtin_inff());
}
__device__ __forceinline__ float clamp11(float v) {
    return __builtin_amdgcn_fmed3f(v, -1.0f, 1.0f);
}

// A-fragment (16x32 bf16) per-lane layout: lanes 0-15 hold K=[0..7],[16..23],
// lanes 16-31 hold K=[8..15],[24..31] of their row. Two contiguous 16B chunks.
__device__ __forceinline__ v16bf load_afrag_lds(const unsigned short* rowp, int kc, int lh) {
    V16U t;
    t.u[0] = *reinterpret_cast<const uint4*>(rowp + kc + lh * 8);
    t.u[1] = *reinterpret_cast<const uint4*>(rowp + kc + 16 + lh * 8);
    return t.v;
}

__device__ __forceinline__ v16bf load_bfrag(const unsigned short* p) {
    // pre-swizzled: 32 contiguous bytes per lane, lane-major blocks of 1KB
    return *reinterpret_cast<const v16bf*>(p);
}

__device__ __forceinline__ v8f wmma_bf16(v16bf a, v16bf b, v8f c) {
    return __builtin_amdgcn_wmma_f32_16x16x32_bf16(false, a, false, b, (short)0, c, false, false);
}

// relu + bf16-convert 8 accumulator rows, store to LDS column (stride HPAD).
// Pairs adjacent rows so the backend emits one v_cvt_pk_bf16_f32 per two rows
// (low half -> ds_store_b16, high half -> ds_store_b16_d16_hi).
__device__ __forceinline__ void store_relu_col(unsigned short* hp, v8f c) {
    #pragma unroll
    for (int i = 0; i < 8; i += 2) {
        v2bf p;
        p.x = f2bf(relu1(c[i]));
        p.y = f2bf(relu1(c[i + 1]));
        unsigned u = __builtin_bit_cast(unsigned, p);
        hp[(i)     * HPAD] = (unsigned short)u;
        hp[(i + 1) * HPAD] = (unsigned short)(u >> 16);
    }
}

// ---------------------------------------------------------------------------
// Prep kernels: swizzle weights into per-lane B-fragment order (bf16)
// packed[((nt*ksteps+ks)*32+lane)*16+kk] = W[(ks*32+8*(lane>>4)+(kk&7)+16*(kk>>3))*Wcols + nt*16+(lane&15)]
// ---------------------------------------------------------------------------
__global__ void pack_weights(const float* __restrict__ W, unsigned short* __restrict__ dst,
                             int ksteps, int Wcols) {
    int idx  = blockIdx.x * blockDim.x + threadIdx.x;
    int kk   = idx & 15;
    int lane = (idx >> 4) & 31;
    int blk  = idx >> 9;
    int ks   = blk % ksteps;
    int nt   = blk / ksteps;
    int k    = ks * 32 + ((lane >> 4) << 3) + (kk & 7) + ((kk >> 3) << 4);
    int n    = nt * 16 + (lane & 15);
    dst[idx] = f2bf_bits(W[k * Wcols + n]);
}

// W3B = W3[256,64] @ Bmat[64,16], packed directly (ksteps=8, ntiles=1)
__global__ void pack_w3b(const float* __restrict__ W3, const float* __restrict__ Bm,
                         unsigned short* __restrict__ dst) {
    int idx  = blockIdx.x * blockDim.x + threadIdx.x;   // 4096 total
    int kk   = idx & 15;
    int lane = (idx >> 4) & 31;
    int ks   = idx >> 9;
    int k    = ks * 32 + ((lane >> 4) << 3) + (kk & 7) + ((kk >> 3) << 4);
    int n    = lane & 15;
    float acc = 0.f;
    #pragma unroll 8
    for (int s = 0; s < STATE_DIM; ++s) acc += W3[k * STATE_DIM + s] * Bm[s * CONTROL_DIM + n];
    dst[idx] = f2bf_bits(acc);
}

// b1eff = b1 + t * W1[64,:];  b3b = b3 @ Bmat
__global__ void prep_bias(const float* __restrict__ t, const float* __restrict__ W1,
                          const float* __restrict__ b1, const float* __restrict__ b3,
                          const float* __restrict__ Bm,
                          float* __restrict__ b1eff, float* __restrict__ b3b) {
    int i = threadIdx.x;
    b1eff[i] = b1[i] + t[0] * W1[STATE_DIM * HIDDEN + i];   // row 64 of W1
    if (i < CONTROL_DIM) {
        float a = 0.f;
        for (int s = 0; s < STATE_DIM; ++s) a += b3[s] * Bm[s * CONTROL_DIM + i];
        b3b[i] = a;
    }
}

// ---------------------------------------------------------------------------
// Fused main kernel: x -> h1 -> h2 -> pB -> u.  One wave owns 16 rows.
// ---------------------------------------------------------------------------
__global__ __launch_bounds__(THREADS)
void oc_fused(const float* __restrict__ x,
              const unsigned short* __restrict__ W1p,
              const unsigned short* __restrict__ W2p,
              const unsigned short* __restrict__ W3bp,
              const float* __restrict__ b1eff,
              const float* __restrict__ b2,
              const float* __restrict__ b3b,
              float* __restrict__ out) {
    extern __shared__ unsigned short sH[];              // [2][MTILE*HPAD] bf16
    unsigned short* sH1 = sH;
    unsigned short* sH2 = sH + MTILE * HPAD;

    const int tid  = threadIdx.x;
    const int wave = tid >> 5;
    const int lane = tid & 31;
    const int l15  = lane & 15;
    const int lh   = lane >> 4;
    const int rowBase = blockIdx.x * MTILE + wave * 16;

    // ---- Layer 1 A fragments straight from global x (f32 -> bf16) ----
    v16bf a1[2];
    #pragma unroll
    for (int ks = 0; ks < 2; ++ks) {
        const float* p0 = x + (size_t)(rowBase + l15) * STATE_DIM + ks * 32 + lh * 8;
        const float* p1 = p0 + 16;
        float4 f0 = *(const float4*)p0, f1 = *(const float4*)(p0 + 4);
        float4 f2 = *(const float4*)p1, f3 = *(const float4*)(p1 + 4);
        v16bf a;
        a[0] = f2bf(f0.x); a[1] = f2bf(f0.y); a[2]  = f2bf(f0.z); a[3]  = f2bf(f0.w);
        a[4] = f2bf(f1.x); a[5] = f2bf(f1.y); a[6]  = f2bf(f1.z); a[7]  = f2bf(f1.w);
        a[8] = f2bf(f2.x); a[9] = f2bf(f2.y); a[10] = f2bf(f2.z); a[11] = f2bf(f2.w);
        a[12]= f2bf(f3.x); a[13]= f2bf(f3.y); a[14] = f2bf(f3.z); a[15] = f2bf(f3.w);
        a1[ks] = a;
    }

    const unsigned short* hrow1 = sH1 + (size_t)(wave * 16 + l15) * HPAD;
    const unsigned short* hrow2 = sH2 + (size_t)(wave * 16 + l15) * HPAD;

    // ---- Layer 1: h1 = relu(x @ W1a + b1eff), K=64 (2 ksteps), N=256 ----
    #pragma unroll 2
    for (int nt = 0; nt < 16; ++nt) {
        float bias = b1eff[nt * 16 + l15];
        v8f c = {bias, bias, bias, bias, bias, bias, bias, bias};
        #pragma unroll
        for (int ks = 0; ks < 2; ++ks) {
            v16bf b = load_bfrag(W1p + ((nt * 2 + ks) * 32 + lane) * 16);
            c = wmma_bf16(a1[ks], b, c);
        }
        store_relu_col(sH1 + (size_t)(wave * 16 + lh * 8) * HPAD + nt * 16 + l15, c);
    }

    // ---- Layer 2 A fragments from LDS (same wave -> LDS in-order, no barrier) ----
    v16bf a2[8];
    #pragma unroll
    for (int ks = 0; ks < 8; ++ks) a2[ks] = load_afrag_lds(hrow1, ks * 32, lh);

    // ---- Layer 2: h2 = relu(h1 @ W2 + b2), K=256 (8 ksteps), N=256 ----
    #pragma unroll 2
    for (int nt = 0; nt < 16; ++nt) {
        float bias = b2[nt * 16 + l15];
        v8f c = {bias, bias, bias, bias, bias, bias, bias, bias};
        #pragma unroll
        for (int ks = 0; ks < 8; ++ks) {
            v16bf b = load_bfrag(W2p + ((nt * 8 + ks) * 32 + lane) * 16);
            c = wmma_bf16(a2[ks], b, c);
        }
        store_relu_col(sH2 + (size_t)(wave * 16 + lh * 8) * HPAD + nt * 16 + l15, c);
    }

    // ---- Layer 3: pB = h2 @ (W3@Bmat) + b3@Bmat, K=256, N=16 ----
    v16bf a3[8];
    #pragma unroll
    for (int ks = 0; ks < 8; ++ks) a3[ks] = load_afrag_lds(hrow2, ks * 32, lh);

    {
        float bias = b3b[l15];
        v8f c = {bias, bias, bias, bias, bias, bias, bias, bias};
        #pragma unroll
        for (int ks = 0; ks < 8; ++ks) {
            v16bf b = load_bfrag(W3bp + (ks * 32 + lane) * 16);
            c = wmma_bf16(a3[ks], b, c);
        }
        // ---- Fixed point: limit of u <- clamp(u - 0.1*(u + pB)) is clamp(-pB),
        //      then the 6 tracked refinement steps (no-ops at the fixed point,
        //      kept for fidelity to the training path). ----
        float* op = out + (size_t)(rowBase + lh * 8) * CONTROL_DIM + l15;
        #pragma unroll
        for (int i = 0; i < 8; ++i) {
            float pb = c[i];
            float u  = clamp11(-pb);
            #pragma unroll
            for (int it = 0; it < 6; ++it)
                u = clamp11(u - 0.1f * (u + pb));
            op[i * CONTROL_DIM] = u;
        }
    }
}

// ---------------------------------------------------------------------------
// Launch
// ---------------------------------------------------------------------------
extern "C" void kernel_launch(void* const* d_in, const int* in_sizes, int n_in,
                              void* d_out, int out_size, void* d_ws, size_t ws_size,
                              hipStream_t stream) {
    const float* x  = (const float*)d_in[0];
    const float* t  = (const float*)d_in[1];
    const float* W1 = (const float*)d_in[2];
    const float* b1 = (const float*)d_in[3];
    const float* W2 = (const float*)d_in[4];
    const float* b2 = (const float*)d_in[5];
    const float* W3 = (const float*)d_in[6];
    const float* b3 = (const float*)d_in[7];
    const float* Bm = (const float*)d_in[8];
    float* out = (float*)d_out;

    const int B = in_sizes[0] / STATE_DIM;   // 131072

    // Workspace layout (all offsets 32B-aligned): ~170 KB
    char* ws = (char*)d_ws;
    unsigned short* W1p  = (unsigned short*)(ws + 0);        //  64*256 bf16 = 32 KB
    unsigned short* W2p  = (unsigned short*)(ws + 32768);    // 256*256 bf16 = 128 KB
    unsigned short* W3bp = (unsigned short*)(ws + 163840);   // 256*16  bf16 =   8 KB
    float*          b1ef = (float*)(ws + 172032);            // 256 f32
    float*          b3b  = (float*)(ws + 173056);            // 16 f32

    pack_weights<<<(64 * 256) / 256, 256, 0, stream>>>(W1, W1p, 2, HIDDEN);
    pack_weights<<<(256 * 256) / 256, 256, 0, stream>>>(W2, W2p, 8, HIDDEN);
    pack_w3b<<<(256 * 16) / 256, 256, 0, stream>>>(W3, Bm, W3bp);
    prep_bias<<<1, 256, 0, stream>>>(t, W1, b1, b3, Bm, b1ef, b3b);

    size_t lds_bytes = (size_t)2 * MTILE * HPAD * sizeof(unsigned short);  // 132 KB
    oc_fused<<<B / MTILE, THREADS, lds_bytes, stream>>>(x, W1p, W2p, W3bp, b1ef, b2, b3b, out);
}
